// MultiHeadAttention_24232205484049
// MI455X (gfx1250) — matrix operand order repaired and checked by reading, not compiled, not run on hardware
//
#include <hip/hip_runtime.h>
#include <hip/hip_bf16.h>
#include <math.h>

// ---------------------------------------------------------------------------
// MHA for MI455X (gfx1250, wave32, WMMA bf16 16x16x32, fp32 accumulate).
// Two-pass column-softmax (softmax over QUERY axis, faithful to reference),
// scores recomputed (never materialized; avoids 536 MB of HBM traffic).
// All B-fragments read contiguous memory via pre-transposed weights /
// transposed V.
// Round 3: ping-pong double buffering (2x unrolled stages) so the register
// rotation of round 2 (v_dual_mov tails) disappears; loads for stage p+1 are
// issued into the alternate buffer while stage p's WMMAs drain partial
// loadcnt.
// ---------------------------------------------------------------------------

namespace {

constexpr int kH  = 16;
constexpr int kDK = 64;
constexpr int kDM = 1024;
constexpr int kB  = 2;
constexpr int kS  = 2048;
constexpr int kM  = kB * kS;              // 4096 token rows
constexpr float kInvSqrtDk = 0.125f;      // 1/sqrt(64)

typedef __attribute__((ext_vector_type(16))) __bf16 v16bf;
typedef __attribute__((ext_vector_type(8)))  __bf16 v8bf;
typedef __attribute__((ext_vector_type(8)))  float  v8f;

__device__ __forceinline__ __bf16 f2bf(float f) {
  union { float f; unsigned u; } v; v.f = f;
  unsigned r = v.u + 0x7fffu + ((v.u >> 16) & 1u);   // round-to-nearest-even
  unsigned short h = (unsigned short)(r >> 16);
  __bf16 o; __builtin_memcpy(&o, &h, 2); return o;
}

__device__ __forceinline__ v16bf frag_cat(v8bf lo, v8bf hi) {
  v16bf r;
#pragma unroll
  for (int i = 0; i < 8; ++i) { r[i] = lo[i]; r[i + 8] = hi[i]; }
  return r;
}

// A-fragment: 16x32 bf16 tile, row-major source with stride lda (elements).
// ISA layout: lane L holds row M=L&15; khi=(L>>4)*8; e<8 -> K=khi+e,
// e>=8 -> K=16+khi+(e-8).  Two 16B loads per lane.
__device__ __forceinline__ v16bf load_a_frag(const __bf16* __restrict__ A,
                                             int lda, int lane) {
  const int row = lane & 15;
  const int khi = (lane >> 4) * 8;
  const __bf16* p = A + (size_t)row * lda + khi;
  return frag_cat(*(const v8bf*)p, *(const v8bf*)(p + 16));
}

// B-fragment: 32x16 bf16 from "transposed" storage Bt[n][k] (k contiguous,
// stride ldb). ISA layout: lane L holds col N=L&15; khi=(L>>4)*8; e<8 -> K=khi+e,
// e>=8 -> K=16+khi+(e-8).
__device__ __forceinline__ v16bf load_b_fragT(const __bf16* __restrict__ Bt,
                                              int ldb, int lane) {
  const int col = lane & 15;
  const int khi = (lane >> 4) * 8;
  const __bf16* p = Bt + (size_t)col * ldb + khi;
  return frag_cat(*(const v8bf*)p, *(const v8bf*)(p + 16));
}

__device__ __forceinline__ v8f wmma_bf16(v16bf a, v16bf b, v8f c) {
  return __builtin_amdgcn_wmma_f32_16x16x32_bf16(
      /*neg_a=*/false, a, /*neg_b=*/false, b,
      /*c_mod=*/(short)0, c, /*reuse_a=*/false, /*reuse_b=*/false);
}

__device__ __forceinline__ v8f v8f_zero() {
  v8f z = {0.f, 0.f, 0.f, 0.f, 0.f, 0.f, 0.f, 0.f};
  return z;
}

} // namespace

// --------------------------- precision / layout prep -----------------------

__global__ void k_cvt_bf16(const float* __restrict__ src,
                           __bf16* __restrict__ dst, int n) {
  for (int i = blockIdx.x * blockDim.x + threadIdx.x; i < n;
       i += gridDim.x * blockDim.x)
    dst[i] = f2bf(src[i]);
}

// src: [batch][rows][cols] f32  ->  dst: [batch][cols][rows] bf16
__global__ void k_transpose_cvt(const float* __restrict__ src,
                                __bf16* __restrict__ dst, int rows, int cols) {
  const size_t base = (size_t)blockIdx.y * rows * cols;
  const int n = rows * cols;
  for (int i = blockIdx.x * blockDim.x + threadIdx.x; i < n;
       i += gridDim.x * blockDim.x) {
    const int r = i / cols, c = i - r * cols;
    dst[base + (size_t)c * rows + r] = f2bf(src[base + i]);
  }
}

// --------------------------- QKV projection --------------------------------
// grid: (kM/128, kH, 3 {q,k,v}), block 128 (4 waves), wave tile = 32x64.
// Ping-pong double buffering: stage p computes from buffer p while issuing
// loads for chunk kk+32 into buffer p^1 (no register rotation copies).
__global__ void __launch_bounds__(128)
k_qkv_proj(const __bf16* __restrict__ xb,
           const __bf16* __restrict__ wqT, const __bf16* __restrict__ wkT,
           const __bf16* __restrict__ wvT,
           const float* __restrict__ bq, const float* __restrict__ bk,
           const float* __restrict__ bv,
           __bf16* __restrict__ qb, __bf16* __restrict__ kb,
           __bf16* __restrict__ vt) {
  const int lane = threadIdx.x & 31;
  const int wave = threadIdx.x >> 5;
  const int m0 = blockIdx.x * 128 + wave * 32;
  const int h = blockIdx.y;
  const int which = blockIdx.z;

  const __bf16* wT =
      (which == 0 ? wqT : which == 1 ? wkT : wvT) + (size_t)h * kDK * kDM;
  const float* bias = (which == 0 ? bq : which == 1 ? bk : bv) + h * kDK;

  v8f acc[2][4];
#pragma unroll
  for (int g = 0; g < 2; ++g)
#pragma unroll
    for (int n = 0; n < 4; ++n) acc[g][n] = v8f_zero();

  v16bf a[2][2], bfr[2][4];   // [buffer][fragment]
#pragma unroll
  for (int g = 0; g < 2; ++g)
    a[0][g] = load_a_frag(xb + (size_t)(m0 + g * 16) * kDM, kDM, lane);
#pragma unroll
  for (int n = 0; n < 4; ++n)
    bfr[0][n] = load_b_fragT(wT + (size_t)(n * 16) * kDM, kDM, lane);

  for (int kk = 0; kk < kDM; kk += 64) {
#pragma unroll
    for (int st = 0; st < 2; ++st) {
      const int cur = st, nxt = st ^ 1;
      const int kn = (kk + (st + 1) * 32) & (kDM - 1);   // wrap: no tail branch
#pragma unroll
      for (int g = 0; g < 2; ++g)
        a[nxt][g] =
            load_a_frag(xb + (size_t)(m0 + g * 16) * kDM + kn, kDM, lane);
#pragma unroll
      for (int n = 0; n < 4; ++n)
        bfr[nxt][n] =
            load_b_fragT(wT + (size_t)(n * 16) * kDM + kn, kDM, lane);

#pragma unroll
      for (int g = 0; g < 2; ++g)
#pragma unroll
        for (int n = 0; n < 4; ++n)
          acc[g][n] = wmma_bf16(a[cur][g], bfr[cur][n], acc[g][n]);
    }
  }

  const int col = lane & 15;
  const int rbase = (lane >> 4) * 8;
#pragma unroll
  for (int g = 0; g < 2; ++g) {
#pragma unroll
    for (int n = 0; n < 4; ++n) {
      const int cg = n * 16 + col;
      const float bv_ = bias[cg];
#pragma unroll
      for (int r = 0; r < 8; ++r) {
        const int m = m0 + g * 16 + rbase + r;
        const int bidx = m >> 11;          // / kS
        const int s = m & (kS - 1);
        const size_t bh = (size_t)(bidx * kH + h);
        const float val = acc[g][n][r] + bv_;
        if (which == 0)
          qb[(bh * kS + s) * kDK + cg] = f2bf(val * kInvSqrtDk);  // fold 1/sqrt(dk)
        else if (which == 1)
          kb[(bh * kS + s) * kDK + cg] = f2bf(val);
        else
          vt[(bh * kDK + cg) * kS + s] = f2bf(val);               // V transposed
      }
    }
  }
}

// --------------------------- pass 1: column stats ---------------------------
// m[t] = max_s scores[s,t];  l[t] = sum_s exp(scores[s,t]-m[t])
// grid: (B*H, kS/64), block 128; each wave owns 16 key columns, streams Q
// with ping-pong buffered A-fragments.
__global__ void __launch_bounds__(128)
k_attn_stats(const __bf16* __restrict__ qb, const __bf16* __restrict__ kb,
             float2* __restrict__ ml) {
  const int lane = threadIdx.x & 31;
  const int wave = threadIdx.x >> 5;
  const int bh = blockIdx.x;
  const int t0 = blockIdx.y * 64 + wave * 16;
  const __bf16* Q = qb + (size_t)bh * kS * kDK;
  const __bf16* K = kb + (size_t)bh * kS * kDK;

  // B = K^T tile (fixed per wave): Bt storage is K row-major itself.
  const v16bf bk0 = load_b_fragT(K + (size_t)t0 * kDK + 0, kDK, lane);
  const v16bf bk1 = load_b_fragT(K + (size_t)t0 * kDK + 32, kDK, lane);

  v16bf aq[2][2];
  aq[0][0] = load_a_frag(Q + 0, kDK, lane);
  aq[0][1] = load_a_frag(Q + 32, kDK, lane);

  float m_run = -3.0e38f, l_run = 0.f;
  for (int s0 = 0; s0 < kS; s0 += 32) {
#pragma unroll
    for (int st = 0; st < 2; ++st) {
      const int cur = st, nxt = st ^ 1;
      const int sn = (s0 + (st + 1) * 16) & (kS - 1);
      __builtin_prefetch(Q + (size_t)((sn + 16) & (kS - 1)) * kDK, 0, 0);
      aq[nxt][0] = load_a_frag(Q + (size_t)sn * kDK + 0, kDK, lane);
      aq[nxt][1] = load_a_frag(Q + (size_t)sn * kDK + 32, kDK, lane);

      v8f c = v8f_zero();
      c = wmma_bf16(aq[cur][0], bk0, c);
      c = wmma_bf16(aq[cur][1], bk1, c);

      // C lane holds column t=t0+(lane&15), 8 query rows. Column reduce =
      // 8 in-lane + one xor-16 shuffle (lane and lane^16 share a column).
      float tm = c[0];
#pragma unroll
      for (int r = 1; r < 8; ++r) tm = fmaxf(tm, c[r]);
      tm = fmaxf(tm, __shfl_xor(tm, 16, 32));
      const float nm = fmaxf(m_run, tm);
      float ts = 0.f;
#pragma unroll
      for (int r = 0; r < 8; ++r) ts += __expf(c[r] - nm);
      ts += __shfl_xor(ts, 16, 32);
      l_run = l_run * __expf(m_run - nm) + ts;   // online column softmax update
      m_run = nm;
    }
  }
  if (lane < 16) {
    float2 v; v.x = m_run; v.y = 1.0f / l_run;
    ml[(size_t)bh * kS + t0 + lane] = v;
  }
}

// --------------------------- pass 2: apply ----------------------------------
// out[s,:] = sum_t exp(scores[s,t]-m[t]) * (1/l[t]) * V[t,:]
// Computes scores^T so the score C-fragment (lane=s col, VGPR=t row) maps
// 1:1 onto the P A-fragment (lane=s row, elements=t) with no shuffles.
// K A-fragments and V B-fragments ping-pong buffered over the t loop.
__global__ void __launch_bounds__(128)
k_attn_apply(const __bf16* __restrict__ qb, const __bf16* __restrict__ kb,
             const __bf16* __restrict__ vt, const float2* __restrict__ ml,
             __bf16* __restrict__ cat) {
  const int lane = threadIdx.x & 31;
  const int wave = threadIdx.x >> 5;
  const int bh = blockIdx.x;
  const int b = bh >> 4;   // / kH
  const int h = bh & 15;
  const int s0 = blockIdx.y * 64 + wave * 16;

  const __bf16* Q = qb + (size_t)bh * kS * kDK;
  const __bf16* K = kb + (size_t)bh * kS * kDK;
  const __bf16* V = vt + (size_t)bh * kDK * kS;       // [k][t]
  const float2* mlp = ml + (size_t)bh * kS;

  // B = Q^T for scores^T (fixed per wave).
  const v16bf bq0 = load_b_fragT(Q + (size_t)s0 * kDK + 0, kDK, lane);
  const v16bf bq1 = load_b_fragT(Q + (size_t)s0 * kDK + 32, kDK, lane);

  v8f acc[4];
#pragma unroll
  for (int n = 0; n < 4; ++n) acc[n] = v8f_zero();

  const int khi8 = (lane >> 4) * 8;

  // Ping-pong buffers: K rows (4 A-frags) and V columns (4 B-frags).
  v16bf ka[2][4], vb[2][4];
#pragma unroll
  for (int i = 0; i < 4; ++i)
    ka[0][i] = load_a_frag(K + (size_t)((i >> 1) * 16) * kDK + (i & 1) * 32,
                           kDK, lane);
#pragma unroll
  for (int n = 0; n < 4; ++n)
    vb[0][n] = load_b_fragT(V + (size_t)(n * 16) * kS, kS, lane);

  for (int tt = 0; tt < kS; tt += 64) {
#pragma unroll
    for (int st = 0; st < 2; ++st) {
      const int cur = st, nxt = st ^ 1;
      const int t0 = tt + st * 32;
      const int tn = (t0 + 32) & (kS - 1);
      __builtin_prefetch(K + (size_t)((t0 + 64) & (kS - 1)) * kDK, 0, 0);

#pragma unroll
      for (int i = 0; i < 4; ++i)
        ka[nxt][i] = load_a_frag(
            K + (size_t)(tn + (i >> 1) * 16) * kDK + (i & 1) * 32, kDK, lane);
#pragma unroll
      for (int n = 0; n < 4; ++n)
        vb[nxt][n] = load_b_fragT(V + (size_t)(n * 16) * kS + tn, kS, lane);

      // scores^T tiles: A = K rows (t), B = Q^T cols (s).
      v8f c0 = v8f_zero(), c1 = v8f_zero();
      c0 = wmma_bf16(ka[cur][0], bq0, c0);
      c0 = wmma_bf16(ka[cur][1], bq1, c0);
      c1 = wmma_bf16(ka[cur][2], bq0, c1);
      c1 = wmma_bf16(ka[cur][3], bq1, c1);

      // P A-fragment: element e<8 comes from c0[e] (t=t0+khi8+e),
      // e>=8 from c1[e-8] (t=t0+16+khi8+(e-8)); lane's s row matches already.
      v16bf pa;
#pragma unroll
      for (int e = 0; e < 8; ++e) {
        const float2 ml0 = mlp[t0 + khi8 + e];
        pa[e] = f2bf(__expf(c0[e] - ml0.x) * ml0.y);
        const float2 ml1 = mlp[t0 + 16 + khi8 + e];
        pa[8 + e] = f2bf(__expf(c1[e] - ml1.x) * ml1.y);
      }

      // out += P(16x32) x V(32x64); B-frags contiguous thanks to Vt layout.
#pragma unroll
      for (int n = 0; n < 4; ++n) acc[n] = wmma_bf16(pa, vb[cur][n], acc[n]);
    }
  }

  // Write concatenated-heads activation [b][s][h*64+k] as bf16 for out-proj.
  const int col = lane & 15;
#pragma unroll
  for (int n = 0; n < 4; ++n) {
    const int cg = h * kDK + n * 16 + col;
#pragma unroll
    for (int r = 0; r < 8; ++r) {
      const int s = s0 + khi8 + r;
      cat[((size_t)b * kS + s) * kDM + cg] = f2bf(acc[n][r]);
    }
  }
}

// --------------------------- output projection ------------------------------
// grid: (kM/128, kDM/64), block 128, wave tile = 32x64, ping-pong buffered.
__global__ void __launch_bounds__(128)
k_out_proj(const __bf16* __restrict__ cat, const __bf16* __restrict__ wcT,
           const float* __restrict__ bc, float* __restrict__ out) {
  const int lane = threadIdx.x & 31;
  const int wave = threadIdx.x >> 5;
  const int m0 = blockIdx.x * 128 + wave * 32;
  const int nb = blockIdx.y * 64;

  v8f acc[2][4];
#pragma unroll
  for (int g = 0; g < 2; ++g)
#pragma unroll
    for (int n = 0; n < 4; ++n) acc[g][n] = v8f_zero();

  v16bf a[2][2], bfr[2][4];
#pragma unroll
  for (int g = 0; g < 2; ++g)
    a[0][g] = load_a_frag(cat + (size_t)(m0 + g * 16) * kDM, kDM, lane);
#pragma unroll
  for (int n = 0; n < 4; ++n)
    bfr[0][n] = load_b_fragT(wcT + (size_t)(nb + n * 16) * kDM, kDM, lane);

  for (int kk = 0; kk < kDM; kk += 64) {
#pragma unroll
    for (int st = 0; st < 2; ++st) {
      const int cur = st, nxt = st ^ 1;
      const int kn = (kk + (st + 1) * 32) & (kDM - 1);
#pragma unroll
      for (int g = 0; g < 2; ++g)
        a[nxt][g] =
            load_a_frag(cat + (size_t)(m0 + g * 16) * kDM + kn, kDM, lane);
#pragma unroll
      for (int n = 0; n < 4; ++n)
        bfr[nxt][n] =
            load_b_fragT(wcT + (size_t)(nb + n * 16) * kDM + kn, kDM, lane);

#pragma unroll
      for (int g = 0; g < 2; ++g)
#pragma unroll
        for (int n = 0; n < 4; ++n)
          acc[g][n] = wmma_bf16(a[cur][g], bfr[cur][n], acc[g][n]);
    }
  }

  const int col = lane & 15;
  const int rbase = (lane >> 4) * 8;
#pragma unroll
  for (int g = 0; g < 2; ++g) {
#pragma unroll
    for (int n = 0; n < 4; ++n) {
      const int cg = nb + n * 16 + col;
      const float bias = bc[cg];
#pragma unroll
      for (int r = 0; r < 8; ++r)
        out[(size_t)(m0 + g * 16 + rbase + r) * kDM + cg] = acc[g][n][r] + bias;
    }
  }
}

// --------------------------- host launcher ----------------------------------

extern "C" void kernel_launch(void* const* d_in, const int* in_sizes, int n_in,
                              void* d_out, int out_size, void* d_ws,
                              size_t ws_size, hipStream_t stream) {
  (void)in_sizes; (void)n_in; (void)out_size; (void)ws_size;

  const float* x  = (const float*)d_in[0];
  const float* Wq = (const float*)d_in[1];
  const float* bq = (const float*)d_in[2];
  const float* Wk = (const float*)d_in[3];
  const float* bk = (const float*)d_in[4];
  const float* Wv = (const float*)d_in[5];
  const float* bv = (const float*)d_in[6];
  const float* Wc = (const float*)d_in[7];
  const float* bc = (const float*)d_in[8];
  float* out = (float*)d_out;

  char* ws = (char*)d_ws;
  size_t off = 0;
  auto take = [&](size_t bytes) -> char* {
    char* p = ws + off;
    off += (bytes + 255) & ~(size_t)255;
    return p;
  };
  __bf16* xb   = (__bf16*)take((size_t)kM * kDM * 2);            // 8 MB
  __bf16* wqT  = (__bf16*)take((size_t)kH * kDK * kDM * 2);      // 2 MB
  __bf16* wkT  = (__bf16*)take((size_t)kH * kDK * kDM * 2);
  __bf16* wvT  = (__bf16*)take((size_t)kH * kDK * kDM * 2);
  __bf16* wcT  = (__bf16*)take((size_t)kDM * kDM * 2);           // 2 MB
  __bf16* qb   = (__bf16*)take((size_t)kB * kH * kS * kDK * 2);  // 8 MB
  __bf16* kbuf = (__bf16*)take((size_t)kB * kH * kS * kDK * 2);
  __bf16* vtb  = (__bf16*)take((size_t)kB * kH * kS * kDK * 2);
  float2* mlb  = (float2*)take((size_t)kB * kH * kS * sizeof(float2));
  __bf16* catb = (__bf16*)take((size_t)kM * kDM * 2);            // 8 MB

  // Precision/layout prep: bf16 everywhere, weights pre-transposed so every
  // WMMA B-fragment is two global_load_b128 per lane.
  k_cvt_bf16<<<dim3(2048), dim3(256), 0, stream>>>(x, xb, kM * kDM);
  k_transpose_cvt<<<dim3(64, kH), dim3(256), 0, stream>>>(Wq, wqT, kDM, kDK);
  k_transpose_cvt<<<dim3(64, kH), dim3(256), 0, stream>>>(Wk, wkT, kDM, kDK);
  k_transpose_cvt<<<dim3(64, kH), dim3(256), 0, stream>>>(Wv, wvT, kDM, kDK);
  k_transpose_cvt<<<dim3(1024, 1), dim3(256), 0, stream>>>(Wc, wcT, kDM, kDM);

  k_qkv_proj<<<dim3(kM / 128, kH, 3), dim3(128), 0, stream>>>(
      xb, wqT, wkT, wvT, bq, bk, bv, qb, kbuf, vtb);

  k_attn_stats<<<dim3(kB * kH, kS / 64), dim3(128), 0, stream>>>(qb, kbuf, mlb);

  k_attn_apply<<<dim3(kB * kH, kS / 64), dim3(128), 0, stream>>>(
      qb, kbuf, vtb, mlb, catb);

  k_out_proj<<<dim3(kM / 128, kDM / 64), dim3(128), 0, stream>>>(
      catb, wcT, bc, out);
}
